// LTCModel_68023692034275
// MI455X (gfx1250) — matrix operand order repaired and checked by reading, not compiled, not run on hardware
//
#include <hip/hip_runtime.h>

typedef __bf16 bf16;
typedef __attribute__((ext_vector_type(16))) __bf16 v16bf;
typedef __attribute__((ext_vector_type(8)))  __bf16 v8bf;
typedef __attribute__((ext_vector_type(8)))  float  v8f;

#define B_N   512
#define W_N   252
#define D_N   128
#define H_N   512
#define OUT_N 500
#define ROWS  (B_N * W_N)   // 129024

// workspace layout (bytes), all 256-aligned
#define OFF_U      ((size_t)0)           // ROWS*512 bf16      = 132,120,576
#define OFF_WCOMB  ((size_t)132120576)   // 1024*1024 bf16     =   2,097,152
#define OFF_WIN    ((size_t)134217728)   // 512*128 bf16       =     131,072
#define OFF_WOUT   ((size_t)134348800)   // 512*512 bf16       =     524,288
#define OFF_H32    ((size_t)134873088)   // 2 * 512*512 f32    =   2,097,152
#define OFF_HBF    ((size_t)136970240)   // 2 * 512*512 bf16   =   1,048,576
// total ~138 MB

// ---- WMMA fragment helpers (CDNA5 16x16x32 bf16 layouts, wave32) ----------
// A (16x32, MxK): lane l holds row (l&15); 8 bf16 at K=(l>>4)*8 and 8 at K=16+(l>>4)*8
__device__ __forceinline__ v16bf load_a_frag(const bf16* base, int stride, int lane) {
    const bf16* p = base + (lane & 15) * stride + ((lane >> 4) << 3);
    v8bf lo = *(const v8bf*)(p);
    v8bf hi = *(const v8bf*)(p + 16);
    return __builtin_shufflevector(lo, hi, 0,1,2,3,4,5,6,7,8,9,10,11,12,13,14,15);
}
// B (32x16, KxN) from transposed storage BT[n][k]: lane l holds col (l&15),
// 8 bf16 at K=(l>>4)*8 and 8 at K=16+(l>>4)*8
__device__ __forceinline__ v16bf load_b_frag(const bf16* base, int stride, int lane) {
    const bf16* p = base + (lane & 15) * stride + ((lane >> 4) << 3);
    v8bf lo = *(const v8bf*)(p);
    v8bf hi = *(const v8bf*)(p + 16);
    return __builtin_shufflevector(lo, hi, 0,1,2,3,4,5,6,7,8,9,10,11,12,13,14,15);
}
__device__ __forceinline__ v8f wmma_bf16(v16bf a, v16bf b, v8f c) {
    return __builtin_amdgcn_wmma_f32_16x16x32_bf16(false, a, false, b, (short)0, c,
                                                   false, false);
}

// ---- CDNA5 async global->LDS copy (ASYNCcnt-tracked), 16B per lane --------
__device__ __forceinline__ void async_copy_b128(bf16* lds_dst, const bf16* gsrc) {
    // LDS aperture maps addr[31:0] directly -> low 32 bits of the generic
    // pointer are the wave-relative LDS byte address the instruction wants.
    unsigned loff = (unsigned)(uintptr_t)lds_dst;
    asm volatile("global_load_async_to_lds_b128 %0, %1, off"
                 :: "v"(loff), "v"(gsrc) : "memory");
}
__device__ __forceinline__ void wait_async0() {
#if __has_builtin(__builtin_amdgcn_s_wait_asynccnt)
    __builtin_amdgcn_s_wait_asynccnt(0);
#else
    asm volatile("s_wait_asynccnt 0" ::: "memory");
#endif
}

// ---- weight packing: bf16 transposed row-major ----------------------------
__global__ void __launch_bounds__(256)
prep_weights(const float* W_in, const float* W_x, const float* W_h,
             const float* W_tx, const float* W_th, const float* W_out,
             bf16* winT, bf16* wcombT, bf16* woutT) {
    int i = blockIdx.x * blockDim.x + threadIdx.x;
    if (i < 512 * 128) {                       // winT[n][k] = W_in[k][n]
        int n = i >> 7, k = i & 127;
        winT[i] = (bf16)W_in[k * 512 + n];
        return;
    }
    int j = i - 512 * 128;
    if (j < 1024 * 1024) {                     // wcombT[n][k], K = [u_t | h]
        int n = j >> 10, k = j & 1023;
        const float* Wtop = (n < 512) ? W_x : W_tx;
        const float* Wbot = (n < 512) ? W_h : W_th;
        int nc = n & 511;
        float v = (k < 512) ? Wtop[k * 512 + nc] : Wbot[(k - 512) * 512 + nc];
        wcombT[j] = (bf16)v;
        return;
    }
    int m = j - 1024 * 1024;
    if (m < 512 * 512) {                       // woutT[n][k], pad 500->512
        int n = m >> 9, k = m & 511;
        woutT[m] = (n < OUT_N) ? (bf16)W_out[k * OUT_N + n] : (bf16)0.f;
    }
}

__global__ void __launch_bounds__(256) init_h(float* h32, bf16* hbf) {
    int i = blockIdx.x * blockDim.x + threadIdx.x;  // 262144 threads
    h32[i] = 0.f;
    hbf[i] = (bf16)0.f;
}

// ---- phase 1: u = tanh(LN(x @ W_in + b_in)) over all B*W rows -------------
__global__ void __launch_bounds__(256)
proj_kernel(const float* __restrict__ x, const float* __restrict__ b_in,
            const float* __restrict__ ln_g, const float* __restrict__ ln_b,
            const bf16* __restrict__ winT, bf16* __restrict__ u_out) {
    __shared__ bf16  lds_a[16 * 136];   // 16 rows x 128 K (padded)
    __shared__ float lds_z[16 * 520];   // 16 rows x 512 cols (padded)
    __shared__ float psum[256], psq[256];
    __shared__ float rmean[16], rinv[16];

    int tid = threadIdx.x, lane = tid & 31, wave = tid >> 5;
    long row0 = (long)blockIdx.x * 16;

    {   // stage x tile, convert fp32 -> bf16
        int r = tid >> 4, c = (tid & 15) * 8;
        const float* src = x + (row0 + r) * D_N + c;
        bf16* dst = lds_a + r * 136 + c;
        #pragma unroll
        for (int e = 0; e < 8; ++e) dst[e] = (bf16)src[e];
    }
    __syncthreads();

    v8f acc[4] = {};
    #pragma unroll
    for (int kt = 0; kt < 4; ++kt) {
        int K0 = kt * 32;
        v16bf a = load_a_frag(lds_a + K0, 136, lane);
        #pragma unroll
        for (int j = 0; j < 4; ++j) {
            int nt = wave * 4 + j;                     // N subtile 0..31
            v16bf b = load_b_frag(winT + (size_t)(nt * 16) * D_N + K0, D_N, lane);
            acc[j] = wmma_bf16(a, b, acc[j]);
        }
    }
    #pragma unroll
    for (int j = 0; j < 4; ++j) {                      // z -> LDS (+bias)
        int col = (wave * 4 + j) * 16 + (lane & 15);
        float bias = b_in[col];
        #pragma unroll
        for (int r = 0; r < 8; ++r) {
            int mrow = r + ((lane >> 4) << 3);
            lds_z[mrow * 520 + col] = acc[j][r] + bias;
        }
    }
    __syncthreads();
    {   // LayerNorm stats: 16 threads/row, 32 elems each
        int r = tid >> 4, seg = tid & 15;
        float s = 0.f, q = 0.f;
        const float* zp = lds_z + r * 520 + seg * 32;
        #pragma unroll
        for (int e = 0; e < 32; ++e) { float v = zp[e]; s += v; q += v * v; }
        psum[tid] = s; psq[tid] = q;
    }
    __syncthreads();
    if (tid < 16) {
        float s = 0.f, q = 0.f;
        #pragma unroll
        for (int e = 0; e < 16; ++e) { s += psum[tid * 16 + e]; q += psq[tid * 16 + e]; }
        float mean = s * (1.f / 512.f);
        float var  = q * (1.f / 512.f) - mean * mean;
        rmean[tid] = mean;
        rinv[tid]  = rsqrtf(var + 1e-5f);
    }
    __syncthreads();
    #pragma unroll 4
    for (int i = 0; i < 32; ++i) {                     // normalize + tanh -> u (bf16)
        int e = tid + i * 256;
        int r = e >> 9, col = e & 511;
        float z = lds_z[r * 520 + col];
        float un = (z - rmean[r]) * rinv[r] * ln_g[col] + ln_b[col];
        u_out[(size_t)(row0 + r) * H_N + col] = (bf16)tanhf(un);
    }
}

// ---- phase 2: one recurrence step: [u_t | h] @ Wcomb, fused LTC update ----
// Double-buffered LDS, async global->LDS staging, one barrier per K-tile.
__global__ void __launch_bounds__(256)
ltc_step(const bf16* __restrict__ u, const bf16* __restrict__ wcombT,
         const bf16* __restrict__ h_bf_in, const float* __restrict__ h32_in,
         bf16* __restrict__ h_bf_out, float* __restrict__ h32_out,
         const float* __restrict__ b_f, const float* __restrict__ b_tau,
         const float* __restrict__ delta_t, float* __restrict__ tau_out, int t) {
    __shared__ bf16 lds_a[2][64 * 40];   // 64 batch rows x 32 K, double-buffered
    __shared__ bf16 lds_b[2][64 * 40];   // 32 f-cols + 32 tau-cols x 32 K
    int tid = threadIdx.x, lane = tid & 31, wave = tid >> 5;
    int mBlk = blockIdx.x;            // rows mBlk*64
    int cBlk = blockIdx.y;            // f cols cBlk*32 (tau cols +512)
    int mSub = wave & 3, cSel = wave >> 2;

    v8f accF = {}, accT = {};
    int srow = tid >> 2, schunk = (tid & 3) * 8;   // each thread: one 16B chunk

    // per-thread source addresses (A side picks u_t or h depending on K)
    int bRow = mBlk * 64 + srow;
    const bf16* uRow = u + (size_t)(bRow * W_N + t) * H_N + schunk;
    const bf16* hRow = h_bf_in + (size_t)bRow * H_N + schunk;
    int nRow = (srow < 32) ? (cBlk * 32 + srow) : (512 + cBlk * 32 + (srow - 32));
    const bf16* wRow = wcombT + (size_t)nRow * 1024 + schunk;
    bf16* aDst = &lds_a[0][srow * 40 + schunk];
    bf16* bDst = &lds_b[0][srow * 40 + schunk];
    const int BUFSZ = 64 * 40;

    // prologue: stage K-tile 0 into buffer 0
    async_copy_b128(aDst, uRow);
    async_copy_b128(bDst, wRow);

    for (int kt = 0; kt < 32; ++kt) {
        int cur = kt & 1;
        wait_async0();           // own async copies for tile kt complete
        __syncthreads();         // -> whole workgroup's tile kt visible
        if (kt + 1 < 32) {       // stage tile kt+1 into the other buffer
            int K0 = (kt + 1) * 32;
            const bf16* asrc = (K0 < 512) ? (uRow + K0) : (hRow + (K0 - 512));
            async_copy_b128(aDst + (cur ^ 1) * BUFSZ, asrc);
            async_copy_b128(bDst + (cur ^ 1) * BUFSZ, wRow + K0);
        }
        const bf16* aBuf = lds_a[cur];
        const bf16* bBuf = lds_b[cur];
        v16bf a  = load_a_frag(aBuf + mSub * 16 * 40, 40, lane);
        v16bf bF = load_b_frag(bBuf + cSel * 16 * 40, 40, lane);
        v16bf bT = load_b_frag(bBuf + (32 + cSel * 16) * 40, 40, lane);
        accF = wmma_bf16(a, bF, accF);
        accT = wmma_bf16(a, bT, accT);
        // no second barrier needed: tile kt+1 overwrites the buffer that was
        // last read at iteration kt-1, and every wave's ds reads of that
        // buffer completed (s_wait_dscnt before its WMMA) before it passed
        // this iteration's barrier.
    }

    int col = cBlk * 32 + cSel * 16 + (lane & 15);     // [0,512)
    float biasF = b_f[col], biasT = b_tau[col];
    int rbase = mBlk * 64 + mSub * 16 + ((lane >> 4) << 3);
    #pragma unroll
    for (int r = 0; r < 8; ++r) {
        int row = rbase + r;
        float f   = tanhf(accF[r] + biasF);
        float sig = 1.f / (1.f + expf(-(accT[r] + biasT)));
        float tau = 0.1f + 9.9f * sig;
        float dt  = delta_t[row * W_N + t];
        float alpha = expf(-dt / tau);
        float hnew  = alpha * h32_in[row * H_N + col] + (1.f - alpha) * f;
        h32_out[row * H_N + col]  = hnew;
        h_bf_out[row * H_N + col] = (bf16)hnew;
        tau_out[row * H_N + col]  = tau;   // last step leaves tau_dist
    }
}

// ---- phase 3: scores = h @ W_out + b_out ----------------------------------
__global__ void __launch_bounds__(256)
scores_kernel(const bf16* __restrict__ h_bf, const bf16* __restrict__ woutT,
              const float* __restrict__ b_out, float* __restrict__ out) {
    __shared__ bf16 lds_a[64 * 40];
    __shared__ bf16 lds_b[32 * 40];
    int tid = threadIdx.x, lane = tid & 31, wave = tid >> 5;
    int mBlk = blockIdx.x, cBlk = blockIdx.y;
    int mSub = wave & 3, cSel = wave >> 2;
    v8f acc = {};
    int srow = tid >> 2, schunk = (tid & 3) * 8;

    for (int kt = 0; kt < 16; ++kt) {
        int K0 = kt * 32;
        *(v8bf*)(lds_a + srow * 40 + schunk) =
            *(const v8bf*)(h_bf + (size_t)(mBlk * 64 + srow) * H_N + K0 + schunk);
        if (srow < 32)
            *(v8bf*)(lds_b + srow * 40 + schunk) =
                *(const v8bf*)(woutT + (size_t)(cBlk * 32 + srow) * 512 + K0 + schunk);
        __syncthreads();
        v16bf a = load_a_frag(lds_a + mSub * 16 * 40, 40, lane);
        v16bf b = load_b_frag(lds_b + cSel * 16 * 40, 40, lane);
        acc = wmma_bf16(a, b, acc);
        __syncthreads();
    }
    int col = cBlk * 32 + cSel * 16 + (lane & 15);
    if (col < OUT_N) {
        float bias = b_out[col];
        int rbase = mBlk * 64 + mSub * 16 + ((lane >> 4) << 3);
        #pragma unroll
        for (int r = 0; r < 8; ++r)
            out[(rbase + r) * OUT_N + col] = acc[r] + bias;
    }
}

extern "C" void kernel_launch(void* const* d_in, const int* in_sizes, int n_in,
                              void* d_out, int out_size, void* d_ws, size_t ws_size,
                              hipStream_t stream) {
    (void)in_sizes; (void)n_in; (void)out_size; (void)ws_size;
    const float* x     = (const float*)d_in[0];
    const float* dt    = (const float*)d_in[1];
    const float* W_in  = (const float*)d_in[2];
    const float* b_in  = (const float*)d_in[3];
    const float* ln_g  = (const float*)d_in[4];
    const float* ln_b  = (const float*)d_in[5];
    const float* W_x   = (const float*)d_in[6];
    const float* W_h   = (const float*)d_in[7];
    const float* b_f   = (const float*)d_in[8];
    const float* W_tx  = (const float*)d_in[9];
    const float* W_th  = (const float*)d_in[10];
    const float* b_tau = (const float*)d_in[11];
    const float* W_out = (const float*)d_in[12];
    const float* b_out = (const float*)d_in[13];

    char* ws = (char*)d_ws;
    bf16*  u_buf  = (bf16*)(ws + OFF_U);
    bf16*  wcombT = (bf16*)(ws + OFF_WCOMB);
    bf16*  winT   = (bf16*)(ws + OFF_WIN);
    bf16*  woutT  = (bf16*)(ws + OFF_WOUT);
    float* h32    = (float*)(ws + OFF_H32);   // 2 ping-pong buffers
    bf16*  hbf    = (bf16*)(ws + OFF_HBF);    // 2 ping-pong buffers

    float* scores  = (float*)d_out;
    float* tau_out = scores + (size_t)B_N * OUT_N;

    prep_weights<<<5376, 256, 0, stream>>>(W_in, W_x, W_h, W_tx, W_th, W_out,
                                           winT, wcombT, woutT);
    init_h<<<1024, 256, 0, stream>>>(h32, hbf);
    proj_kernel<<<ROWS / 16, 256, 0, stream>>>(x, b_in, ln_g, ln_b, winT, u_buf);

    const size_t HS = (size_t)B_N * H_N;
    for (int t = 0; t < W_N; ++t) {
        int rd = t & 1, wr = rd ^ 1;
        ltc_step<<<dim3(8, 16), 256, 0, stream>>>(
            u_buf, wcombT, hbf + (size_t)rd * HS, h32 + (size_t)rd * HS,
            hbf + (size_t)wr * HS, h32 + (size_t)wr * HS,
            b_f, b_tau, dt, tau_out, t);
    }
    // t=251 wrote buffer index 0
    scores_kernel<<<dim3(8, 16), 256, 0, stream>>>(hbf, woutT, b_out, scores);
}
